// MusicGen_15814069584073
// MI455X (gfx1250) — compile-verified, matrix operand order checked by reading, not run on hardware
//
#include <hip/hip_runtime.h>
#include <math.h>

// ---------------- types ----------------
typedef __bf16 bf16;
typedef __attribute__((ext_vector_type(16))) __bf16 v16bf;
typedef __attribute__((ext_vector_type(8)))  __bf16 v8bf;
typedef __attribute__((ext_vector_type(8)))  float  v8f;
typedef __attribute__((ext_vector_type(4)))  float  f32x4;

static __device__ __forceinline__ bf16 bz() {
    return __builtin_bit_cast(bf16, (unsigned short)0);
}
// fp32 -> bf16, round-to-nearest-even (bit manipulation)
static __device__ __forceinline__ bf16 f2bf(float f) {
    unsigned u = __builtin_bit_cast(unsigned, f);
    unsigned r = u + 0x7FFFu + ((u >> 16) & 1u);
    unsigned short s = (unsigned short)(r >> 16);
    return __builtin_bit_cast(bf16, s);
}
static __device__ __forceinline__ v8f v8f_zero() {
    v8f z = {0.f,0.f,0.f,0.f,0.f,0.f,0.f,0.f};
    return z;
}
static __device__ __forceinline__ v16bf cat16(v8bf lo, v8bf hi) {
    return __builtin_shufflevector(lo, hi, 0,1,2,3,4,5,6,7,8,9,10,11,12,13,14,15);
}
static __device__ __forceinline__ float sigm(float x) { return 1.f / (1.f + __expf(-x)); }

// ---- CDNA5 async global->LDS copy (ASYNCcnt path, ISA 10.x / 15.18.3) ----
// LDS dest operand is the byte offset within the workgroup LDS allocation;
// generic pointers to LDS have that offset in addr[31:0] (aperture rule).
static __device__ __forceinline__ void async_b128(void* lds, const void* g) {
    unsigned off = (unsigned)(unsigned long long)lds;
    asm volatile("global_load_async_to_lds_b128 %0, %1, off"
                 :: "v"(off), "v"(g) : "memory");
}
static __device__ __forceinline__ void wait_async_le4() {
    asm volatile("s_wait_asynccnt 0x4" ::: "memory");
}
static __device__ __forceinline__ void wait_async_0() {
    asm volatile("s_wait_asynccnt 0x0" ::: "memory");
}

// problem constants
#define VV 32000
#define II 512
#define HH 1024
#define BB 8
#define LL 128
#define TT 256
#define SS 383            // L + T - 1
#define ROWS (BB * SS)    // 3064
#define ROWS_PAD 3072     // padded to block-tile multiple (pad rows = garbage, masked)
#define SEQ_PAD 3200      // Ob/seq1 row allocation (covers S-GEMM N-tile overrun)
#define G3H  (3 * HH)     // 3072

// ================= generic bf16 WMMA GEMM =================
// C[M,N](f32, per-batch contiguous) = act( A[M,K](bf16,row-major) * B + bias[N] )
// BT=true : B stored (N,K) row-major (A @ W^T). K%32==0 at all BT call sites and
//           A/B allocations are row-padded -> guard-free double-buffered
//           global_load_async_to_lds_b128 pipeline (ASYNCcnt).
// BT=false: B stored (K,N) row-major (ctx = A @ O, K=383) -> guarded elementwise.
// Block tile 128x128x32, 256 threads = 8 waves (2 M x 4 N), wave tile 64x32.
template<bool BT, bool TANH>
__global__ __launch_bounds__(256)
void wmma_gemm(const bf16* __restrict__ Ag, const bf16* __restrict__ Bg,
               const float* __restrict__ bias, float* __restrict__ Cg,
               int M, int N, int K, long sA, long sB, long sC)
{
    __shared__ __align__(16) bf16 As[2][128][32];
    __shared__ __align__(16) bf16 Bs[2][128][32];   // always stored [n][k]

    const bf16* A = Ag + (long)blockIdx.z * sA;
    const bf16* B = Bg + (long)blockIdx.z * sB;
    float*      C = Cg + (long)blockIdx.z * sC;

    const int tid  = threadIdx.x;
    const int lane = tid & 31;
    const int wave = tid >> 5;
    const int wm   = wave & 1;       // M sub-tile: 64*wm
    const int wn   = wave >> 1;      // N sub-tile: 32*wn
    const int half = lane >> 4;      // ISA: K-group select
    const int rc   = lane & 15;      // row (A) / col (B,C)

    const int m_base = blockIdx.y * 128;
    const int n_base = blockIdx.x * 128;

    v8f acc[4][2];
    #pragma unroll
    for (int i = 0; i < 4; ++i)
        #pragma unroll
        for (int j = 0; j < 2; ++j) acc[i][j] = v8f_zero();

    const int nK = (K + 31) >> 5;

    // each thread owns 2 A-segments + 2 B-segments (8 bf16 = 16B each) per tile
    const int r0  = tid >> 2,         sg0 = tid & 3;         // seg index tid
    const int r1  = (tid + 256) >> 2, sg1 = tid & 3;         // seg index tid+256
    auto issue_tile = [&](int buf, int kt) {
        const int k0 = kt << 5;
        async_b128(&As[buf][r0][sg0 * 8], A + (long)(m_base + r0) * K + k0 + sg0 * 8);
        async_b128(&Bs[buf][r0][sg0 * 8], B + (long)(n_base + r0) * K + k0 + sg0 * 8);
        async_b128(&As[buf][r1][sg1 * 8], A + (long)(m_base + r1) * K + k0 + sg1 * 8);
        async_b128(&Bs[buf][r1][sg1 * 8], B + (long)(n_base + r1) * K + k0 + sg1 * 8);
    };

    if (BT) issue_tile(0, 0);

    for (int kt = 0; kt < nK; ++kt) {
        const int cur = kt & 1;
        if (BT) {
            if (kt + 1 < nK) { issue_tile(cur ^ 1, kt + 1); wait_async_le4(); }
            else             { wait_async_0(); }
        } else {
            const int k0 = kt << 5;
            for (int i = tid; i < 128 * 32; i += 256) {
                int r = i >> 5, k = i & 31;
                int gm = m_base + r, gk = k0 + k;
                As[0][r][k] = (gm < M && gk < K) ? A[(long)gm * K + gk] : bz();
            }
            for (int i = tid; i < 128 * 32; i += 256) {
                int r = i >> 5, k = i & 31;      // r = n index within tile
                int gn = n_base + r, gk = k0 + k;
                Bs[0][r][k] = (gn < N && gk < K) ? B[(long)gk * N + gn] : bz();
            }
        }
        __syncthreads();

        const int rb = BT ? cur : 0;
        // ---- fragments per ISA 7.12.2; 16B ds_load_b128 pairs ----
        v16bf bfrag[2];
        #pragma unroll
        for (int nf = 0; nf < 2; ++nf) {
            const bf16* Brow = &Bs[rb][wn * 32 + nf * 16 + rc][0];
            v8bf lo = *(const v8bf*)(Brow + half * 8);
            v8bf hi = *(const v8bf*)(Brow + 16 + half * 8);
            bfrag[nf] = cat16(lo, hi);
        }
        #pragma unroll
        for (int mf = 0; mf < 4; ++mf) {
            const bf16* Arow = &As[rb][wm * 64 + mf * 16 + rc][0];
            v8bf lo = *(const v8bf*)(Arow + half * 8);
            v8bf hi = *(const v8bf*)(Arow + 16 + half * 8);
            v16bf af = cat16(lo, hi);
            #pragma unroll
            for (int nf = 0; nf < 2; ++nf) {
                acc[mf][nf] = __builtin_amdgcn_wmma_f32_16x16x32_bf16(
                    false, af, false, bfrag[nf], (short)0, acc[mf][nf], false, false);
            }
        }
        __syncthreads();   // fences buffer reuse for iteration kt+1's async issue
    }

    // ---- epilogue (C layout: VGPR r -> M = r + 8*half; N = lane&15) ----
    #pragma unroll
    for (int mf = 0; mf < 4; ++mf) {
        #pragma unroll
        for (int nf = 0; nf < 2; ++nf) {
            int n = n_base + wn * 32 + nf * 16 + rc;
            if (n >= N) continue;
            float bv = bias ? bias[n] : 0.f;
            #pragma unroll
            for (int r = 0; r < 8; ++r) {
                int m = m_base + wm * 64 + mf * 16 + half * 8 + r;
                if (m < M) {
                    float v = acc[mf][nf][r] + bv;
                    if (TANH) v = tanhf(v);
                    C[(long)m * N + n] = v;
                }
            }
        }
    }
}

// ================= GRU recurrent step: gh = h @ Whh^T + bhh =================
// h is PADDED to 16 rows (rows 8..15 stay zero) -> branchless b128 loads.
// M=16 (8 real), N=3072, K=1024. 8 waves/block, 1 n-frag each, grid=24.
__global__ __launch_bounds__(256)
void gru_gh(const bf16* __restrict__ h, const bf16* __restrict__ Whh,
            const float* __restrict__ bhh, float* __restrict__ gh)
{
    const int lane = threadIdx.x & 31;
    const int wave = threadIdx.x >> 5;
    const int half = lane >> 4;
    const int rc   = lane & 15;
    const int n0   = blockIdx.x * 128 + wave * 16;

    const bf16* hrow = h + (long)rc * HH;             // rows 8..15 are zero
    const bf16* wrow = Whh + (long)(n0 + rc) * HH;

    v8f acc = v8f_zero();
    for (int k0 = 0; k0 < HH; k0 += 32) {
        v8bf alo = *(const v8bf*)(hrow + k0 + half * 8);
        v8bf ahi = *(const v8bf*)(hrow + k0 + 16 + half * 8);
        v8bf blo = *(const v8bf*)(wrow + k0 + half * 8);
        v8bf bhi = *(const v8bf*)(wrow + k0 + 16 + half * 8);
        __builtin_prefetch((const void*)(wrow + k0 + 64), 0, 1);
        acc = __builtin_amdgcn_wmma_f32_16x16x32_bf16(
            false, cat16(alo, ahi), false, cat16(blo, bhi), (short)0, acc, false, false);
    }
    const int n = n0 + rc;
    #pragma unroll
    for (int r = 0; r < 8; ++r) {
        int m = half * 8 + r;
        if (m < BB) gh[(long)m * G3H + n] = acc[r] + bhh[n];
    }
}

// ================= GRU gate update (one timestep) =================
__global__ void gru_update(const float* __restrict__ Gi, const float* __restrict__ gh,
                           float* __restrict__ h32, bf16* __restrict__ hbf,
                           bf16* __restrict__ seq, int t)
{
    int idx = blockIdx.x * blockDim.x + threadIdx.x;   // 0..8191
    if (idx >= BB * HH) return;
    int b = idx >> 10, j = idx & (HH - 1);
    const float* gib = Gi + ((long)b * SS + t) * G3H;
    const float* ghb = gh + (long)b * G3H;
    float r = sigm(gib[j] + ghb[j]);
    float z = sigm(gib[HH + j] + ghb[HH + j]);
    float n = tanhf(gib[2 * HH + j] + r * ghb[2 * HH + j]);
    float hnew = (1.f - z) * n + z * h32[idx];
    h32[idx] = hnew;
    bf16 hb = f2bf(hnew);
    hbf[idx] = hb;                                     // rows 0..7 of padded h
    seq[((long)b * SS + t) * HH + j] = hb;
}

// zero h32 (8 rows) and padded hbf (16 rows)
__global__ void zero_h(float* h32, bf16* hbf)
{
    int idx = blockIdx.x * blockDim.x + threadIdx.x;
    if (idx < BB * HH) h32[idx] = 0.f;
    if (idx < 16 * HH) hbf[idx] = bz();
}

// ================= embedding gather -> bf16 (8-wide vectorized) =================
__global__ void embed_gather(const int* __restrict__ input, const int* __restrict__ targets,
                             const float* __restrict__ embed_w, bf16* __restrict__ emb)
{
    long v = (long)blockIdx.x * blockDim.x + threadIdx.x;  // 8-element groups
    if (v >= (long)ROWS * II / 8) return;
    int row = (int)(v / (II / 8)), g = (int)(v % (II / 8));
    int b = row / SS, t = row % SS;
    int tok = (t < LL) ? input[b * LL + t] : targets[b * TT + (t - LL)];
    const f32x4* src = (const f32x4*)(embed_w + (long)tok * II + g * 8);
    f32x4 a = src[0], c = src[1];
    v8bf o;
    #pragma unroll
    for (int i = 0; i < 4; ++i) { o[i] = f2bf(a[i]); o[4 + i] = f2bf(c[i]); }
    *(v8bf*)(emb + (long)row * II + g * 8) = o;
}

// ================= fp32 -> bf16 converter (8-wide; n % 8 == 0) =================
__global__ void cvt_f32_bf16(const float* __restrict__ src, bf16* __restrict__ dst, long n8)
{
    long v = (long)blockIdx.x * blockDim.x + threadIdx.x;
    if (v >= n8) return;
    const f32x4* s = (const f32x4*)(src + v * 8);
    f32x4 a = s[0], c = s[1];
    v8bf o;
    #pragma unroll
    for (int i = 0; i < 4; ++i) { o[i] = f2bf(a[i]); o[4 + i] = f2bf(c[i]); }
    *(v8bf*)(dst + v * 8) = o;
}

// ================= masked softmax row (k < 127+q), emits bf16 A =================
__global__ __launch_bounds__(256)
void softmax_mask(const float* __restrict__ S, bf16* __restrict__ Ab)
{
    int row = blockIdx.x;                 // b*256 + q
    int q = row & (TT - 1);
    int nvalid = (LL - 1) + q;            // valid k in [0, nvalid)
    const float* Sr = S + (long)row * SS;
    bf16* Ar = Ab + (long)row * SS;
    __shared__ float red[256];
    int tid = threadIdx.x;

    float mx = -1e30f;
    for (int k = tid; k < nvalid; k += 256) mx = fmaxf(mx, Sr[k]);
    red[tid] = mx; __syncthreads();
    for (int s = 128; s > 0; s >>= 1) { if (tid < s) red[tid] = fmaxf(red[tid], red[tid + s]); __syncthreads(); }
    mx = red[0]; __syncthreads();

    float sum = 0.f;
    for (int k = tid; k < nvalid; k += 256) sum += __expf(Sr[k] - mx);
    red[tid] = sum; __syncthreads();
    for (int s = 128; s > 0; s >>= 1) { if (tid < s) red[tid] += red[tid + s]; __syncthreads(); }
    float inv = 1.f / red[0];

    for (int k = tid; k < SS; k += 256) {
        float a = (k < nvalid) ? __expf(Sr[k] - mx) * inv : 0.f;
        Ar[k] = f2bf(a);
    }
}

// ================= concat [Q, ctx] -> bf16 =================
__global__ void build_cat(const bf16* __restrict__ O, const float* __restrict__ ctx,
                          bf16* __restrict__ cat)
{
    long idx = (long)blockIdx.x * blockDim.x + threadIdx.x;
    if (idx >= (long)BB * TT * 2 * HH) return;
    int row = (int)(idx >> 11);          // b*256+q
    int j = (int)(idx & (2 * HH - 1));
    int b = row >> 8, q = row & (TT - 1);
    bf16 v;
    if (j < HH) v = O[((long)b * SS + (LL - 1) + q) * HH + j];  // Q = O[:, 127+q]
    else        v = f2bf(ctx[(long)row * HH + (j - HH)]);
    cat[idx] = v;
}

// ================= host orchestration =================
extern "C" void kernel_launch(void* const* d_in, const int* in_sizes, int n_in,
                              void* d_out, int out_size, void* d_ws, size_t ws_size,
                              hipStream_t stream)
{
    const int*   input   = (const int*)  d_in[0];
    const int*   targets = (const int*)  d_in[1];
    const float* embed_w = (const float*)d_in[2];
    const float* Wih0    = (const float*)d_in[3];
    const float* Whh0    = (const float*)d_in[4];
    const float* bih0    = (const float*)d_in[5];
    const float* bhh0    = (const float*)d_in[6];
    const float* Wih1    = (const float*)d_in[7];
    const float* Whh1    = (const float*)d_in[8];
    const float* bih1    = (const float*)d_in[9];
    const float* bhh1    = (const float*)d_in[10];
    const float* Wc      = (const float*)d_in[11];
    const float* bc      = (const float*)d_in[12];
    const float* Wv      = (const float*)d_in[13];
    const float* bv      = (const float*)d_in[14];
    float* logits = (float*)d_out;

    // --- workspace carve-out (~182 MB), 256B aligned; A/B operands row-padded ---
    char* w = (char*)d_ws;
    auto carve = [&](size_t bytes) { char* p = w; w += (bytes + 255) & ~(size_t)255; return p; };
    bf16* wWih0 = (bf16*)carve((size_t)G3H * II * 2);
    bf16* wWhh0 = (bf16*)carve((size_t)G3H * HH * 2);
    bf16* wWih1 = (bf16*)carve((size_t)G3H * HH * 2);
    bf16* wWhh1 = (bf16*)carve((size_t)G3H * HH * 2);
    bf16* wWc   = (bf16*)carve((size_t)HH * 2 * HH * 2);
    bf16* wWv   = (bf16*)carve((size_t)VV * HH * 2);
    bf16* emb   = (bf16*)carve((size_t)ROWS_PAD * II * 2);   // padded rows
    float* Gi   = (float*)carve((size_t)ROWS * G3H * 4);     // reused layer0/layer1
    bf16* seq1  = (bf16*)carve((size_t)SEQ_PAD * HH * 2);    // layer-0 outputs (padded)
    bf16* Ob    = (bf16*)carve((size_t)SEQ_PAD * HH * 2);    // layer-1 outputs (padded)
    float* h32  = (float*)carve((size_t)BB * HH * 4);
    bf16* hbf   = (bf16*)carve((size_t)16 * HH * 2);         // padded to 16 rows
    float* gh   = (float*)carve((size_t)BB * G3H * 4);
    float* Sbuf = (float*)carve((size_t)BB * TT * SS * 4);
    bf16* Ab    = (bf16*)carve((size_t)BB * TT * SS * 2);
    float* ctx  = (float*)carve((size_t)BB * TT * HH * 4);
    bf16* cat   = (bf16*)carve((size_t)BB * TT * 2 * HH * 2);
    float* comb = (float*)carve((size_t)BB * TT * HH * 4);
    bf16* combb = (bf16*)carve((size_t)BB * TT * HH * 2);

    auto cvt = [&](const float* s, bf16* d, long n) {
        long n8 = n / 8;
        cvt_f32_bf16<<<(unsigned)((n8 + 255) / 256), 256, 0, stream>>>(s, d, n8);
    };
    // 1) weights -> bf16
    cvt(Wih0, wWih0, (long)G3H * II);
    cvt(Whh0, wWhh0, (long)G3H * HH);
    cvt(Wih1, wWih1, (long)G3H * HH);
    cvt(Whh1, wWhh1, (long)G3H * HH);
    cvt(Wc,   wWc,   (long)HH * 2 * HH);
    cvt(Wv,   wWv,   (long)VV * HH);

    // 2) embedding gather (8-wide)
    {
        long n = (long)ROWS * II / 8;
        embed_gather<<<(unsigned)((n + 255) / 256), 256, 0, stream>>>(input, targets, embed_w, emb);
    }

    // 3) layer 0: Gi0 = emb @ Wih0^T + bih0   (3064 x 512 -> 3072)
    wmma_gemm<true, false><<<dim3(G3H / 128, (ROWS + 127) / 128, 1), 256, 0, stream>>>(
        emb, wWih0, bih0, Gi, ROWS, G3H, II, 0, 0, 0);

    // 4) layer-0 recurrent scan
    zero_h<<<(16 * HH + 255) / 256, 256, 0, stream>>>(h32, hbf);
    for (int t = 0; t < SS; ++t) {
        gru_gh<<<G3H / 128, 256, 0, stream>>>(hbf, wWhh0, bhh0, gh);
        gru_update<<<(BB * HH + 255) / 256, 256, 0, stream>>>(Gi, gh, h32, hbf, seq1, t);
    }

    // 5) layer 1: Gi1 = seq1 @ Wih1^T + bih1  (3064 x 1024 -> 3072)
    wmma_gemm<true, false><<<dim3(G3H / 128, (ROWS + 127) / 128, 1), 256, 0, stream>>>(
        seq1, wWih1, bih1, Gi, ROWS, G3H, HH, 0, 0, 0);

    // 6) layer-1 recurrent scan -> O
    zero_h<<<(16 * HH + 255) / 256, 256, 0, stream>>>(h32, hbf);
    for (int t = 0; t < SS; ++t) {
        gru_gh<<<G3H / 128, 256, 0, stream>>>(hbf, wWhh1, bhh1, gh);
        gru_update<<<(BB * HH + 255) / 256, 256, 0, stream>>>(Gi, gh, h32, hbf, Ob, t);
    }

    // 7) S = Q @ O^T  (batched; Q = O rows 127..382; async pipeline path)
    wmma_gemm<true, false><<<dim3((SS + 127) / 128, (TT + 127) / 128, BB), 256, 0, stream>>>(
        Ob + (long)(LL - 1) * HH, Ob, nullptr, Sbuf,
        TT, SS, HH, (long)SS * HH, (long)SS * HH, (long)TT * SS);

    // 8) masked softmax -> bf16 A (masked lanes exactly 0)
    softmax_mask<<<BB * TT, 256, 0, stream>>>(Sbuf, Ab);

    // 9) ctx = A @ O  (batched, non-transposed B, K=383 zero-padded in LDS)
    wmma_gemm<false, false><<<dim3(HH / 128, (TT + 127) / 128, BB), 256, 0, stream>>>(
        Ab, Ob, nullptr, ctx,
        TT, HH, SS, (long)TT * SS, (long)SS * HH, (long)TT * HH);

    // 10) cat = [Q, ctx] in bf16
    {
        long n = (long)BB * TT * 2 * HH;
        build_cat<<<(unsigned)((n + 255) / 256), 256, 0, stream>>>(Ob, ctx, cat);
    }

    // 11) comb = tanh(cat @ Wc^T + bc)
    wmma_gemm<true, true><<<dim3(HH / 128, (BB * TT) / 128, 1), 256, 0, stream>>>(
        cat, wWc, bc, comb, BB * TT, HH, 2 * HH, 0, 0, 0);
    cvt(comb, combb, (long)BB * TT * HH);

    // 12) logits = comb @ Wv^T + bv  (dominant GEMM: 2048 x 1024 -> 32000)
    wmma_gemm<true, false><<<dim3(VV / 128, (BB * TT) / 128, 1), 256, 0, stream>>>(
        combb, wWv, bv, logits, BB * TT, VV, HH, 0, 0, 0);
}
